// Attention_74388833566897
// MI455X (gfx1250) — matrix operand order repaired and checked by reading, not compiled, not run on hardware
//
#include <hip/hip_runtime.h>
#include <hip/hip_bf16.h>

// ---------------------------------------------------------------------------
// MLA attention forward for MI455X (gfx1250), bf16 WMMA pipeline.
// D = A(M,K) * B(N,K)^T GEMMs + flash-attention, all through
// v_wmma_f32_16x16x32_bf16. 128x128 GEMM tiles (1.5 LDS loads / WMMA),
// V written pre-transposed so attention staging is all b128.
// ---------------------------------------------------------------------------

typedef __attribute__((ext_vector_type(16))) __bf16 bf16x16;
typedef __attribute__((ext_vector_type(8)))  __bf16 bf16x8;
typedef __attribute__((ext_vector_type(8)))  float  v8f;

union FragBF { bf16x16 v; bf16x8 h[2]; };
union FAcc   { v8f v; float f[8]; };

__device__ __forceinline__ float to_f32(float x)   { return x; }
__device__ __forceinline__ float to_f32(__bf16 x)  { return (float)x; }

__device__ __forceinline__ v8f wmma_bf16(bf16x16 a, bf16x16 b, v8f c) {
    return __builtin_amdgcn_wmma_f32_16x16x32_bf16(
        /*neg_a=*/false, a, /*neg_b=*/false, b,
        /*c_mod=*/(short)0, c, /*reuse_a=*/false, /*reuse_b=*/false);
}

// ---------------------------------------------------------------------------
// Tiled GEMM: D[M,N] = A[M,K] * B[N,K]^T  (row-major A,B).
// Block tile BM x BN, K-step 64; 8 waves; each wave -> (16*WMT) x (16*WNT).
// MODE 0: D row-major (ld = N).  MODE 1: scatter into V^T layout
//   [b, h, d, T] with N==2048 (h*128+d), M==B*T (b*2048+t).
// ---------------------------------------------------------------------------
template<int BM, int BN, int WMT, int WNT, int MODE,
         typename TA, typename TB, typename TC>
__global__ __launch_bounds__(256)
void gemm_abt(const TA* __restrict__ A, const TB* __restrict__ Bm,
              TC* __restrict__ D, int K, int N)
{
    constexpr int LDT = 72;  // 64 + 8 pad; 144B row stride (16B multiple)
    constexpr int WGM = BM / (16 * WMT);   // waves along m
    constexpr int WGN = BN / (16 * WNT);   // waves along n
    static_assert(WGM * WGN == 8, "block must hold 8 waves");

    __shared__ __align__(32) __bf16 As[BM * LDT];
    __shared__ __align__(32) __bf16 Bs[BN * LDT];

    const int t    = threadIdx.x;
    const int wid  = t >> 5;
    const int lane = t & 31;
    const int half = lane >> 4;
    const int lm   = lane & 15;
    const int wm   = wid % WGM;
    const int wn   = wid / WGM;

    const int m0 = blockIdx.y * BM;
    const int n0 = blockIdx.x * BN;

    // cooperative-load decomposition: 64 rows x 64 cols per pass, 16/thr
    const int lrw = t >> 2;
    const int lcg = (t & 3) * 16;

    FAcc acc[WMT][WNT];
#pragma unroll
    for (int mt = 0; mt < WMT; ++mt)
#pragma unroll
        for (int nt = 0; nt < WNT; ++nt)
#pragma unroll
            for (int r = 0; r < 8; ++r) acc[mt][nt].f[r] = 0.0f;

    for (int k0 = 0; k0 < K; k0 += 64) {
#pragma unroll
        for (int rr = 0; rr < BM; rr += 64) {
            const int row = rr + lrw;
            const TA* pa = A + (size_t)(m0 + row) * K + k0 + lcg;
            __bf16* da = &As[row * LDT + lcg];
#pragma unroll
            for (int i = 0; i < 16; ++i) da[i] = (__bf16)to_f32(pa[i]);
        }
#pragma unroll
        for (int rr = 0; rr < BN; rr += 64) {
            const int row = rr + lrw;
            const TB* pb = Bm + (size_t)(n0 + row) * K + k0 + lcg;
            __bf16* db = &Bs[row * LDT + lcg];
#pragma unroll
            for (int i = 0; i < 16; ++i) db[i] = (__bf16)to_f32(pb[i]);
        }
        __syncthreads();
#pragma unroll
        for (int kc = 0; kc < 2; ++kc) {
            const int kb = kc * 32;
            FragBF fa[WMT];
#pragma unroll
            for (int mt = 0; mt < WMT; ++mt) {
                const int row = (wm * WMT + mt) * 16 + lm;
                fa[mt].h[0] = *(const bf16x8*)&As[row * LDT + kb + half * 8];
                fa[mt].h[1] = *(const bf16x8*)&As[row * LDT + kb + 16 + half * 8];
            }
#pragma unroll
            for (int nt = 0; nt < WNT; ++nt) {
                FragBF fb;
                const __bf16* bp = &Bs[((wn * WNT + nt) * 16 + lm) * LDT + kb];
                fb.h[0] = *(const bf16x8*)(bp + half * 8);
                fb.h[1] = *(const bf16x8*)(bp + 16 + half * 8);
#pragma unroll
                for (int mt = 0; mt < WMT; ++mt)
                    acc[mt][nt].v = wmma_bf16(fa[mt].v, fb.v, acc[mt][nt].v);
            }
        }
        __syncthreads();
    }

    // C/D layout: lane holds col n = lane&15; row m = r + 8*(lane>>4)
#pragma unroll
    for (int mt = 0; mt < WMT; ++mt) {
#pragma unroll
        for (int nt = 0; nt < WNT; ++nt) {
#pragma unroll
            for (int r = 0; r < 8; ++r) {
                const int mrow = m0 + (wm * WMT + mt) * 16 + r + 8 * half;
                const int ncol = n0 + (wn * WNT + nt) * 16 + lm;
                if (MODE == 0) {
                    D[(size_t)mrow * N + ncol] = (TC)acc[mt][nt].f[r];
                } else {
                    // V^T: [b, h, d, T]; b = mrow/2048, t = mrow%2048,
                    //      h = ncol/128, d = ncol%128
                    const size_t bb = (size_t)(mrow >> 11);
                    const size_t tt = (size_t)(mrow & 2047);
                    const size_t hh = (size_t)(ncol >> 7);
                    const size_t dd = (size_t)(ncol & 127);
                    D[(((bb << 4) | hh) * 128 + dd) * 2048 + tt] =
                        (TC)acc[mt][nt].f[r];
                }
            }
        }
    }
}

// ---------------------------------------------------------------------------
// RoPE (bf16 in/out, fp32 math).  q_r: (B*T, H*64); k_r: (B*T, 64).
// ---------------------------------------------------------------------------
__global__ void rope_qr_kernel(__bf16* __restrict__ q,
                               const float* __restrict__ c,
                               const float* __restrict__ s)
{
    const int idx = blockIdx.x * blockDim.x + threadIdx.x;  // B*T*H*32
    const int j  = idx & 31;
    const int h  = (idx >> 5) & 15;
    const int bt = idx >> 9;
    const int tt = bt & 2047;
    const float cf = c[tt * 32 + j], sf = s[tt * 32 + j];
    __bf16* p = q + (size_t)bt * 1024 + h * 64 + j * 2;
    const float re = (float)p[0], im = (float)p[1];
    p[0] = (__bf16)(re * cf - im * sf);
    p[1] = (__bf16)(re * sf + im * cf);
}

__global__ void rope_kr_kernel(__bf16* __restrict__ k,
                               const float* __restrict__ c,
                               const float* __restrict__ s)
{
    const int idx = blockIdx.x * blockDim.x + threadIdx.x;  // B*T*32
    const int j  = idx & 31;
    const int bt = idx >> 5;
    const int tt = bt & 2047;
    const float cf = c[tt * 32 + j], sf = s[tt * 32 + j];
    __bf16* p = k + (size_t)bt * 64 + j * 2;
    const float re = (float)p[0], im = (float)p[1];
    p[0] = (__bf16)(re * cf - im * sf);
    p[1] = (__bf16)(re * sf + im * cf);
}

// ---------------------------------------------------------------------------
// Flash attention. Block = 8 waves = 128 q-rows for one (b,h).
// Each wave owns 16 q-rows; K/V streamed in 32-wide tiles through LDS.
// V arrives pre-transposed [b,h,d,T] so all staging is b128.
// d_qk = 192 (128 content + 64 rope), d_v = 128. Causal.
// ---------------------------------------------------------------------------
__global__ __launch_bounds__(256)
void attn_kernel(const __bf16* __restrict__ qc, const __bf16* __restrict__ qr,
                 const __bf16* __restrict__ kc, const __bf16* __restrict__ kr,
                 const __bf16* __restrict__ vt, __bf16* __restrict__ out)
{
    constexpr int   Tn    = 2048;
    constexpr float SCALE = 0.07216878364870322f;  // 1/sqrt(192)
    constexpr float NEGMX = -3.0e38f;
    constexpr int LDK = 200;  // 192 + 8 pad
    constexpr int LDV = 40;   // 32  + 8 pad
    constexpr int LDP = 40;

    __shared__ __align__(32) __bf16 Ks[32 * LDK];        // [kv][d]
    __shared__ __align__(32) __bf16 Vs[128 * LDV];       // [d][kv]
    __shared__ __align__(32) __bf16 Ps[8 * 16 * LDP];    // per-wave P patch

    const int tid  = threadIdx.x;
    const int wid  = tid >> 5;
    const int lane = tid & 31;
    const int half = lane >> 4;
    const int ln   = lane & 15;

    const int qt = blockIdx.x;          // q tile (128 rows)
    const int bh = blockIdx.y;
    const int b  = bh >> 4;
    const int h  = bh & 15;
    const int q0 = qt * 128 + wid * 16;
    const size_t base = (size_t)b * Tn;
    const __bf16* vhead = vt + ((size_t)bh * 128) * Tn;   // [d][T] for this b,h

    // ---- load Q fragments (A layout): 6 k-chunks of 32 over d=0..191 ----
    FragBF qf[6];
    {
        const int qrow = q0 + ln;
        const __bf16* qcrow = qc + (base + qrow) * 2048 + h * 128;
        const __bf16* qrrow = qr + (base + qrow) * 1024 + h * 64;
#pragma unroll
        for (int kf = 0; kf < 6; ++kf) {
            const int dA = kf * 32 + half * 8;
            const int dB = kf * 32 + 16 + half * 8;
            qf[kf].h[0] = *(const bf16x8*)((dA < 128) ? (qcrow + dA) : (qrrow + dA - 128));
            qf[kf].h[1] = *(const bf16x8*)((dB < 128) ? (qcrow + dB) : (qrrow + dB - 128));
        }
    }

    FAcc o[8];
    float mrow[8], lrow[8];
#pragma unroll
    for (int dt = 0; dt < 8; ++dt)
#pragma unroll
        for (int r = 0; r < 8; ++r) o[dt].f[r] = 0.0f;
#pragma unroll
    for (int r = 0; r < 8; ++r) { mrow[r] = NEGMX; lrow[r] = 0.0f; }

    // cooperative staging decomposition (reused each tile)
    const int krow = tid >> 3;          // kv row 0..31 (K tile)
    const int kchk = tid & 7;           // 8-col chunk base (x3 passes)
    const int vd   = tid >> 1;          // d row 0..127 (V tile)
    const int vc   = (tid & 1) * 16;    // kv half

    const int kv_end = (qt + 1) * 128;
    for (int j = 0; j < kv_end; j += 32) {
        // ---- cooperative K/V tile staging, all b128 ----
        {
            const __bf16* kcrow = kc + (base + j + krow) * 2048 + h * 128;
            const __bf16* krrow = kr + (base + j + krow) * 64;
            __bf16* dst = &Ks[krow * LDK];
#pragma unroll
            for (int it = 0; it < 3; ++it) {
                const int c8 = (kchk + 8 * it) * 8;   // 0..184 step 8
                const __bf16* src = (c8 < 128) ? (kcrow + c8) : (krrow + c8 - 128);
                *(bf16x8*)&dst[c8] = *(const bf16x8*)src;
            }
            const __bf16* vsrc = vhead + (size_t)vd * Tn + j + vc;
            *(bf16x8*)&Vs[vd * LDV + vc]     = *(const bf16x8*)vsrc;
            *(bf16x8*)&Vs[vd * LDV + vc + 8] = *(const bf16x8*)(vsrc + 8);
            if (j + 32 < kv_end) {  // prefetch next K row + next V tile -> global_prefetch_b8
                __builtin_prefetch(kcrow + 2048, 0, 0);
                __builtin_prefetch(vsrc + 32, 0, 0);
            }
        }
        __syncthreads();

        // ---- S = Q * K^T : 16 x 32 per wave (2 n-subtiles) ----
        FAcc s[2];
#pragma unroll
        for (int nt = 0; nt < 2; ++nt)
#pragma unroll
            for (int r = 0; r < 8; ++r) s[nt].f[r] = 0.0f;
#pragma unroll
        for (int kf = 0; kf < 6; ++kf) {
#pragma unroll
            for (int nt = 0; nt < 2; ++nt) {
                FragBF kb;
                const __bf16* kp = &Ks[(nt * 16 + ln) * LDK + kf * 32];
                kb.h[0] = *(const bf16x8*)(kp + half * 8);
                kb.h[1] = *(const bf16x8*)(kp + 16 + half * 8);
                s[nt].v = wmma_bf16(qf[kf].v, kb.v, s[nt].v);
            }
        }

        // ---- online softmax (per-row reductions across 16-lane halves) ----
#pragma unroll
        for (int r = 0; r < 8; ++r) {
            const int qidx = q0 + r + 8 * half;
            float s0 = s[0].f[r] * SCALE;
            float s1 = s[1].f[r] * SCALE;
            if (j + ln      > qidx) s0 = NEGMX;
            if (j + 16 + ln > qidx) s1 = NEGMX;
            float tmax = fmaxf(s0, s1);
#pragma unroll
            for (int off = 1; off <= 8; off <<= 1)
                tmax = fmaxf(tmax, __shfl_xor(tmax, off, 32));
            const float mn    = fmaxf(mrow[r], tmax);
            const float alpha = __expf(mrow[r] - mn);
            const float p0 = __expf(s0 - mn);
            const float p1 = __expf(s1 - mn);
            float ps = p0 + p1;
#pragma unroll
            for (int off = 1; off <= 8; off <<= 1)
                ps += __shfl_xor(ps, off, 32);
            mrow[r] = mn;
            lrow[r] = lrow[r] * alpha + ps;
#pragma unroll
            for (int dt = 0; dt < 8; ++dt) o[dt].f[r] *= alpha;
            const int mloc = r + 8 * half;
            Ps[(wid * 16 + mloc) * LDP + ln]      = (__bf16)p0;
            Ps[(wid * 16 + mloc) * LDP + 16 + ln] = (__bf16)p1;
        }
        asm volatile("s_wait_dscnt 0" ::: "memory");  // wave-local P RAW

        // ---- O += P(16x32) * V(32x128) ----
        FragBF pa;
        {
            const __bf16* pp = &Ps[(wid * 16 + ln) * LDP];
            pa.h[0] = *(const bf16x8*)(pp + half * 8);
            pa.h[1] = *(const bf16x8*)(pp + 16 + half * 8);
        }
#pragma unroll
        for (int dt = 0; dt < 8; ++dt) {
            FragBF vb;
            const __bf16* vp = &Vs[(dt * 16 + ln) * LDV];
            vb.h[0] = *(const bf16x8*)(vp + half * 8);
            vb.h[1] = *(const bf16x8*)(vp + 16 + half * 8);
            o[dt].v = wmma_bf16(pa.v, vb.v, o[dt].v);
        }
        __syncthreads();
    }

    // ---- finalize: O /= rowsum; write (b,q,h,d) bf16 ----
#pragma unroll
    for (int r = 0; r < 8; ++r) {
        const int qidx = q0 + r + 8 * half;
        const float inv = 1.0f / lrow[r];
        __bf16* orow = out + (base + qidx) * 2048 + h * 128;
#pragma unroll
        for (int dt = 0; dt < 8; ++dt)
            orow[dt * 16 + ln] = (__bf16)(o[dt].f[r] * inv);
    }
}

// ---------------------------------------------------------------------------
// Host launch
// ---------------------------------------------------------------------------
extern "C" void kernel_launch(void* const* d_in, const int* in_sizes, int n_in,
                              void* d_out, int out_size, void* d_ws, size_t ws_size,
                              hipStream_t stream)
{
    (void)in_sizes; (void)n_in; (void)out_size; (void)ws_size;

    const float* x     = (const float*)d_in[0];
    const float* fcos  = (const float*)d_in[1];
    const float* fsin  = (const float*)d_in[2];
    const float* W_dq  = (const float*)d_in[3];
    const float* W_uq  = (const float*)d_in[4];
    const float* W_dkv = (const float*)d_in[5];
    const float* W_uk  = (const float*)d_in[6];
    const float* W_uv  = (const float*)d_in[7];
    const float* W_qr  = (const float*)d_in[8];
    const float* W_kr  = (const float*)d_in[9];
    const float* W_o   = (const float*)d_in[10];

    const size_t BT = 4096;  // B*T
    char* p = (char*)d_ws;
    auto alloc = [&](size_t bytes) {
        char* r = p;
        p += (bytes + 255) & ~(size_t)255;
        return r;
    };
    __bf16* c_q  = (__bf16*)alloc(BT * 512  * 2);
    __bf16* c_kv = (__bf16*)alloc(BT * 512  * 2);
    __bf16* q_c  = (__bf16*)alloc(BT * 2048 * 2);
    __bf16* k_c  = (__bf16*)alloc(BT * 2048 * 2);
    __bf16* v_t  = (__bf16*)alloc(BT * 2048 * 2);   // [b,h,d,T]
    __bf16* q_r  = (__bf16*)alloc(BT * 1024 * 2);
    __bf16* k_r  = (__bf16*)alloc(BT * 64   * 2);
    __bf16* attn = (__bf16*)alloc(BT * 2048 * 2);

    dim3 blk(256);

    // stage 1: down-projections from x (K=2048)
    gemm_abt<128, 128, 2, 4, 0, float, float, __bf16>
        <<<dim3(512 / 128, BT / 128), blk, 0, stream>>>(x, W_dq,  c_q,  2048, 512);
    gemm_abt<128, 128, 2, 4, 0, float, float, __bf16>
        <<<dim3(512 / 128, BT / 128), blk, 0, stream>>>(x, W_dkv, c_kv, 2048, 512);
    gemm_abt<64, 64, 1, 2, 0, float, float, __bf16>
        <<<dim3(64 / 64, BT / 64), blk, 0, stream>>>(x, W_kr, k_r, 2048, 64);

    // stage 2: up-projections from latents (K=512)
    gemm_abt<128, 128, 2, 4, 0, __bf16, float, __bf16>
        <<<dim3(2048 / 128, BT / 128), blk, 0, stream>>>(c_q,  W_uq, q_c, 512, 2048);
    gemm_abt<128, 128, 2, 4, 0, __bf16, float, __bf16>
        <<<dim3(1024 / 128, BT / 128), blk, 0, stream>>>(c_q,  W_qr, q_r, 512, 1024);
    gemm_abt<128, 128, 2, 4, 0, __bf16, float, __bf16>
        <<<dim3(2048 / 128, BT / 128), blk, 0, stream>>>(c_kv, W_uk, k_c, 512, 2048);
    gemm_abt<128, 128, 2, 4, 1, __bf16, float, __bf16>   // V stored transposed
        <<<dim3(2048 / 128, BT / 128), blk, 0, stream>>>(c_kv, W_uv, v_t, 512, 2048);

    // RoPE
    rope_qr_kernel<<<(BT * 16 * 32) / 256, 256, 0, stream>>>(q_r, fcos, fsin);
    rope_kr_kernel<<<(BT * 32) / 256,      256, 0, stream>>>(k_r, fcos, fsin);

    // flash attention: grid (q-tiles of 128, B*H)
    attn_kernel<<<dim3(2048 / 128, 2 * 16), blk, 0, stream>>>(q_c, q_r, k_c, k_r, v_t, attn);

    // output projection -> fp32 d_out
    gemm_abt<128, 128, 2, 4, 0, __bf16, float, float>
        <<<dim3(2048 / 128, BT / 128), blk, 0, stream>>>(attn, W_o, (float*)d_out, 2048, 2048);
}
